// MoeLayer_45397804319142
// MI455X (gfx1250) — matrix operand order repaired and checked by reading, not compiled, not run on hardware
//
#include <hip/hip_runtime.h>
#include <cstdint>
#include <cstddef>

typedef __attribute__((ext_vector_type(16))) _Float16 v16h;
typedef __attribute__((ext_vector_type(8)))  _Float16 v8h;
typedef __attribute__((ext_vector_type(4)))  _Float16 v4h;
typedef __attribute__((ext_vector_type(8)))  float    v8f;
typedef __attribute__((ext_vector_type(4)))  float    v4f;

constexpr int TOK = 4096;          // B*S tokens
constexpr int HD  = 1024;          // hidden
constexpr int FF  = 2048;          // dff
constexpr int NE  = 8;             // experts
constexpr int KC  = 32;            // K-chunk (one f16 WMMA K)
constexpr int ROWS_CAP = TOK * 2 + NE * 64;  // padded compacted rows

// ---- workspace layout (256B aligned) ----
constexpr size_t OFF_CNT  = 0;
constexpr size_t OFF_OFFS = 256;
constexpr size_t OFF_AE   = 512;
constexpr size_t OFF_AI   = OFF_AE + (size_t)TOK * 2 * sizeof(int);
constexpr size_t OFF_AW   = OFF_AI + (size_t)TOK * 2 * sizeof(int);
constexpr size_t OFF_SLOT = OFF_AW + (size_t)TOK * 2 * sizeof(float);
constexpr size_t SZ_X     = (size_t)TOK * HD * sizeof(_Float16);
constexpr size_t SZ_W     = (size_t)NE * FF * HD * sizeof(_Float16);
constexpr size_t SZ_H     = (size_t)ROWS_CAP * FF * sizeof(_Float16);
constexpr size_t OFF_XH   = OFF_SLOT + (size_t)NE * TOK * sizeof(int);
constexpr size_t OFF_XL   = OFF_XH + SZ_X;
constexpr size_t OFF_W1H  = OFF_XL + SZ_X;
constexpr size_t OFF_W1L  = OFF_W1H + SZ_W;
constexpr size_t OFF_W2H  = OFF_W1L + SZ_W;
constexpr size_t OFF_W2L  = OFF_W2H + SZ_W;
constexpr size_t OFF_W3H  = OFF_W2L + SZ_W;
constexpr size_t OFF_W3L  = OFF_W3H + SZ_W;
constexpr size_t OFF_HH   = OFF_W3L + SZ_W;
constexpr size_t OFF_HL   = OFF_HH + SZ_H;
constexpr size_t OFF_EO   = OFF_HL + SZ_H;

// ---- gfx1250 async global->LDS copy (ASYNCcnt-tracked) ----
// LDS byte address = low 32 bits of the generic shared-aperture address.
__device__ __forceinline__ void async_ld16(void* lds, const void* g) {
  asm volatile("global_load_async_to_lds_b128 %0, %1, off"
               :: "v"((unsigned)(size_t)lds), "v"(g) : "memory");
}
__device__ __forceinline__ void wait_async() {
  asm volatile("s_wait_asynccnt 0" ::: "memory");
}

// XOR-swizzled LDS offset (halves): group g (8 halves) of row r.
// Keeps 16B alignment; 16 rows x one group cover all 16 bank-quads.
__device__ __forceinline__ int swz(int r, int g) {
  return r * 32 + ((g ^ (r & 3) ^ ((r >> 2) & 3)) << 3);
}

// A 16x32 f16 fragment: lane<16 -> M=lane, K=0..15; lane>=16 -> K=16..31
__device__ __forceinline__ v16h frag_a(const _Float16* s, int mb, int lane) {
  const int r = mb + (lane & 15);
  const int g0 = (lane >> 4) << 1;          // 0 or 2
  v8h p0 = *(const v8h*)(s + swz(r, g0));
  v8h p1 = *(const v8h*)(s + swz(r, g0 + 1));
  return __builtin_shufflevector(p0, p1, 0,1,2,3,4,5,6,7,8,9,10,11,12,13,14,15);
}

// B 32x16 f16 fragment: lane<16 -> N=lane, K=0..15; lane>=16 -> K=16..31
__device__ __forceinline__ v16h frag_b(const _Float16* s, int nb, int lane) {
  const int r = nb + (lane & 15);
  const int g0 = (lane >> 4) << 1;          // 0 or 2
  v8h p0 = *(const v8h*)(s + swz(r, g0));
  v8h p1 = *(const v8h*)(s + swz(r, g0 + 1));
  return __builtin_shufflevector(p0, p1, 0,1,2,3,4,5,6,7,8,9,10,11,12,13,14,15);
}

// compensated product: ah*bh + al*bh + ah*bl (f32 accumulate) ~ fp32-quality
__device__ __forceinline__ v8f wmma3(v16h ah, v16h al, v16h bh, v16h bl, v8f acc) {
  acc = __builtin_amdgcn_wmma_f32_16x16x32_f16(false, ah, false, bh, (short)0, acc, false, false);
  acc = __builtin_amdgcn_wmma_f32_16x16x32_f16(false, al, false, bh, (short)0, acc, false, false);
  acc = __builtin_amdgcn_wmma_f32_16x16x32_f16(false, ah, false, bl, (short)0, acc, false, false);
  return acc;
}

// ---------- 0) split f32 -> (f16 hi, f16 lo) ----------
__global__ __launch_bounds__(256) void conv_split(const float* __restrict__ src,
                                                  _Float16* __restrict__ dh,
                                                  _Float16* __restrict__ dl,
                                                  unsigned n4) {
  unsigned i = blockIdx.x * 256u + threadIdx.x;
  if (i >= n4) return;
  v4f v = *(const v4f*)(src + (size_t)i * 4);
  v4h h, l;
#pragma unroll
  for (int j = 0; j < 4; ++j) {
    _Float16 hh = (_Float16)v[j];
    h[j] = hh;
    l[j] = (_Float16)(v[j] - (float)hh);
  }
  *(v4h*)(dh + (size_t)i * 4) = h;
  *(v4h*)(dl + (size_t)i * 4) = l;
}

// ---------- 1) gating: x@Wg.T, top-2 softmax, compacted routing ----------
__global__ __launch_bounds__(256) void moe_gate(
    const float* __restrict__ x, const float* __restrict__ Wg,
    int* __restrict__ cnt, int* __restrict__ assign_e, int* __restrict__ assign_i,
    float* __restrict__ assign_w, int* __restrict__ slots) {
  __shared__ float wgs[NE * HD];
  for (int i = threadIdx.x; i < NE * HD; i += 256) wgs[i] = Wg[i];
  __syncthreads();

  const int wave = threadIdx.x >> 5, lane = threadIdx.x & 31;
  const int t = blockIdx.x * 8 + wave;
  float acc[NE];
#pragma unroll
  for (int e = 0; e < NE; ++e) acc[e] = 0.f;
  const float* xr = x + (size_t)t * HD;
  for (int j = lane; j < HD; j += 32) {
    float xv = xr[j];
#pragma unroll
    for (int e = 0; e < NE; ++e) acc[e] += xv * wgs[e * HD + j];
  }
#pragma unroll
  for (int e = 0; e < NE; ++e)
#pragma unroll
    for (int off = 16; off > 0; off >>= 1)
      acc[e] += __shfl_xor(acc[e], off, 32);

  if (lane == 0) {
    int b0 = 0; float v0 = acc[0];
#pragma unroll
    for (int e = 1; e < NE; ++e) if (acc[e] > v0) { v0 = acc[e]; b0 = e; }
    int bsec = (b0 == 0) ? 1 : 0; float v1 = acc[bsec];
#pragma unroll
    for (int e = 0; e < NE; ++e)
      if (e != b0 && acc[e] > v1) { v1 = acc[e]; bsec = e; }
    float e1 = __expf(v1 - v0);
    float inv = 1.f / (1.f + e1);
    float w0 = inv, w1 = e1 * inv;
    int i0 = atomicAdd(&cnt[b0], 1);
    int i1 = atomicAdd(&cnt[bsec], 1);
    slots[b0 * TOK + i0]   = t;
    slots[bsec * TOK + i1] = t;
    assign_e[t * 2 + 0] = b0;   assign_i[t * 2 + 0] = i0;   assign_w[t * 2 + 0] = w0;
    assign_e[t * 2 + 1] = bsec; assign_i[t * 2 + 1] = i1;   assign_w[t * 2 + 1] = w1;
  }
}

// ---------- 2) pad each expert's rows to 64, compute row offsets ----------
__global__ void moe_prefix(const int* __restrict__ cnt, int* __restrict__ offsets) {
  if (threadIdx.x == 0 && blockIdx.x == 0) {
    int acc = 0;
    for (int e = 0; e < NE; ++e) {
      offsets[e] = acc;
      acc += ((cnt[e] + 63) / 64) * 64;
    }
  }
}

// ---------- 3) FFN1: h = silu(x@W1.T + b1) * (x@W2.T + b2), grouped by expert ----------
__global__ __launch_bounds__(256) void moe_ffn1(
    const _Float16* __restrict__ xh, const _Float16* __restrict__ xl,
    const _Float16* __restrict__ w1h, const _Float16* __restrict__ w1l,
    const _Float16* __restrict__ w2h, const _Float16* __restrict__ w2l,
    const float* __restrict__ b1, const float* __restrict__ b2,
    const int* __restrict__ slots, const int* __restrict__ counts,
    const int* __restrict__ offsets,
    _Float16* __restrict__ hh, _Float16* __restrict__ hl) {
  const int e  = blockIdx.y >> 6;
  const int mt = blockIdx.y & 63;
  const int cnt = counts[e];
  if (mt * 64 >= cnt) return;
  const int nt = blockIdx.x;                 // DFF tile (128 wide)

  // double-buffered f16 tiles (80 KB total)
  __shared__ __align__(16) _Float16 ah_s[2][64 * KC],  al_s[2][64 * KC];
  __shared__ __align__(16) _Float16 b1h_s[2][128 * KC], b1l_s[2][128 * KC];
  __shared__ __align__(16) _Float16 b2h_s[2][128 * KC], b2l_s[2][128 * KC];

  const int tid = threadIdx.x;
  const int wave = tid >> 5, lane = tid & 31;
  const int wm = (wave & 1) * 32;            // wave tile: 32(M) x 32(N)
  const int wn = (wave >> 1) * 32;

  // fill mapping: one 16B group per thread per tile-half
  const int r4 = tid >> 2;                   // 0..63
  const int g4 = tid & 3;                    // 0..3
  const int aoff  = swz(r4, g4);
  const int boff0 = swz(r4, g4);
  const int boff1 = swz(r4 + 64, g4);

  const int idx = mt * 64 + r4;
  const int tok = (idx < cnt) ? slots[e * TOK + idx] : 0;
  const _Float16* xrh = xh + (size_t)tok * HD + g4 * 8;
  const _Float16* xrl = xl + (size_t)tok * HD + g4 * 8;
  const size_t wbase = ((size_t)e * FF + (size_t)nt * 128) * HD;
  const _Float16* w1hb = w1h + wbase + g4 * 8;
  const _Float16* w1lb = w1l + wbase + g4 * 8;
  const _Float16* w2hb = w2h + wbase + g4 * 8;
  const _Float16* w2lb = w2l + wbase + g4 * 8;

  auto fill = [&](int bb, int k) {
    async_ld16(&ah_s[bb][aoff],   xrh + k);
    async_ld16(&al_s[bb][aoff],   xrl + k);
    async_ld16(&b1h_s[bb][boff0], w1hb + (size_t)r4 * HD + k);
    async_ld16(&b1h_s[bb][boff1], w1hb + (size_t)(r4 + 64) * HD + k);
    async_ld16(&b1l_s[bb][boff0], w1lb + (size_t)r4 * HD + k);
    async_ld16(&b1l_s[bb][boff1], w1lb + (size_t)(r4 + 64) * HD + k);
    async_ld16(&b2h_s[bb][boff0], w2hb + (size_t)r4 * HD + k);
    async_ld16(&b2h_s[bb][boff1], w2hb + (size_t)(r4 + 64) * HD + k);
    async_ld16(&b2l_s[bb][boff0], w2lb + (size_t)r4 * HD + k);
    async_ld16(&b2l_s[bb][boff1], w2lb + (size_t)(r4 + 64) * HD + k);
  };

  const v8f vz = {0.f, 0.f, 0.f, 0.f, 0.f, 0.f, 0.f, 0.f};
  v8f accg[2][2], accv[2][2];
#pragma unroll
  for (int i = 0; i < 2; ++i)
#pragma unroll
    for (int j = 0; j < 2; ++j) { accg[i][j] = vz; accv[i][j] = vz; }

  fill(0, 0);
  int bsel = 0;
  for (int k0 = 0; k0 < HD; k0 += KC) {
    wait_async();            // own async fills drained before signaling
    __syncthreads();         // all waves' fills visible -> tiles ready
    if (k0 + KC < HD) fill(bsel ^ 1, k0 + KC);   // stream next chunk under WMMA

    v16h Ah[2], Al[2];
#pragma unroll
    for (int mi = 0; mi < 2; ++mi) {
      Ah[mi] = frag_a(&ah_s[bsel][0], wm + mi * 16, lane);
      Al[mi] = frag_a(&al_s[bsel][0], wm + mi * 16, lane);
    }
#pragma unroll
    for (int ni = 0; ni < 2; ++ni) {
      v16h B1h = frag_b(&b1h_s[bsel][0], wn + ni * 16, lane);
      v16h B1l = frag_b(&b1l_s[bsel][0], wn + ni * 16, lane);
      v16h B2h = frag_b(&b2h_s[bsel][0], wn + ni * 16, lane);
      v16h B2l = frag_b(&b2l_s[bsel][0], wn + ni * 16, lane);
#pragma unroll
      for (int mi = 0; mi < 2; ++mi) {
        accg[mi][ni] = wmma3(Ah[mi], Al[mi], B1h, B1l, accg[mi][ni]);
        accv[mi][ni] = wmma3(Ah[mi], Al[mi], B2h, B2l, accv[mi][ni]);
      }
    }
    bsel ^= 1;
  }

  const int rowbase = offsets[e] + mt * 64;
#pragma unroll
  for (int mi = 0; mi < 2; ++mi)
#pragma unroll
    for (int ni = 0; ni < 2; ++ni) {
      const int n = nt * 128 + wn + ni * 16 + (lane & 15);
      const float bb1 = b1[e * FF + n];
      const float bb2 = b2[e * FF + n];
#pragma unroll
      for (int r = 0; r < 8; ++r) {
        int m = wm + mi * 16 + ((lane >> 4) << 3) + r;
        float g = accg[mi][ni][r] + bb1;
        float v = accv[mi][ni][r] + bb2;
        float h = (g / (1.f + __expf(-g))) * v;     // silu(g) * v
        _Float16 p = (_Float16)h;
        size_t o = (size_t)(rowbase + m) * FF + n;
        hh[o] = p;
        hl[o] = (_Float16)(h - (float)p);
      }
    }
}

// ---------- 4) FFN2: eo = h @ W3.T per expert ----------
__global__ __launch_bounds__(256) void moe_ffn2(
    const _Float16* __restrict__ hh, const _Float16* __restrict__ hl,
    const _Float16* __restrict__ w3h, const _Float16* __restrict__ w3l,
    const int* __restrict__ counts, const int* __restrict__ offsets,
    float* __restrict__ eobuf) {
  const int e  = blockIdx.y >> 6;
  const int mt = blockIdx.y & 63;
  if (mt * 64 >= counts[e]) return;
  const int nt = blockIdx.x;                 // H tile (128 wide)

  __shared__ __align__(16) _Float16 ah_s[2][64 * KC],  al_s[2][64 * KC];
  __shared__ __align__(16) _Float16 bh_s[2][128 * KC], bl_s[2][128 * KC];

  const int tid = threadIdx.x;
  const int wave = tid >> 5, lane = tid & 31;
  const int wm = (wave & 1) * 32;
  const int wn = (wave >> 1) * 32;
  const int r4 = tid >> 2;
  const int g4 = tid & 3;
  const int aoff  = swz(r4, g4);
  const int boff0 = swz(r4, g4);
  const int boff1 = swz(r4 + 64, g4);

  const int rowbase = offsets[e] + mt * 64;
  const _Float16* arh = hh + (size_t)(rowbase + r4) * FF + g4 * 8;
  const _Float16* arl = hl + (size_t)(rowbase + r4) * FF + g4 * 8;
  const size_t wbase = ((size_t)e * HD + (size_t)nt * 128) * FF;
  const _Float16* w3hb = w3h + wbase + g4 * 8;
  const _Float16* w3lb = w3l + wbase + g4 * 8;

  auto fill = [&](int bb, int k) {
    async_ld16(&ah_s[bb][aoff],  arh + k);
    async_ld16(&al_s[bb][aoff],  arl + k);
    async_ld16(&bh_s[bb][boff0], w3hb + (size_t)r4 * FF + k);
    async_ld16(&bh_s[bb][boff1], w3hb + (size_t)(r4 + 64) * FF + k);
    async_ld16(&bl_s[bb][boff0], w3lb + (size_t)r4 * FF + k);
    async_ld16(&bl_s[bb][boff1], w3lb + (size_t)(r4 + 64) * FF + k);
  };

  const v8f vz = {0.f, 0.f, 0.f, 0.f, 0.f, 0.f, 0.f, 0.f};
  v8f acc[2][2];
#pragma unroll
  for (int i = 0; i < 2; ++i)
#pragma unroll
    for (int j = 0; j < 2; ++j) acc[i][j] = vz;

  fill(0, 0);
  int bsel = 0;
  for (int k0 = 0; k0 < FF; k0 += KC) {
    wait_async();
    __syncthreads();
    if (k0 + KC < FF) fill(bsel ^ 1, k0 + KC);

    v16h Ah[2], Al[2];
#pragma unroll
    for (int mi = 0; mi < 2; ++mi) {
      Ah[mi] = frag_a(&ah_s[bsel][0], wm + mi * 16, lane);
      Al[mi] = frag_a(&al_s[bsel][0], wm + mi * 16, lane);
    }
#pragma unroll
    for (int ni = 0; ni < 2; ++ni) {
      v16h Bh = frag_b(&bh_s[bsel][0], wn + ni * 16, lane);
      v16h Bl = frag_b(&bl_s[bsel][0], wn + ni * 16, lane);
#pragma unroll
      for (int mi = 0; mi < 2; ++mi)
        acc[mi][ni] = wmma3(Ah[mi], Al[mi], Bh, Bl, acc[mi][ni]);
    }
    bsel ^= 1;
  }

#pragma unroll
  for (int mi = 0; mi < 2; ++mi)
#pragma unroll
    for (int ni = 0; ni < 2; ++ni) {
      const int n = nt * 128 + wn + ni * 16 + (lane & 15);
#pragma unroll
      for (int r = 0; r < 8; ++r) {
        int m = wm + mi * 16 + ((lane >> 4) << 3) + r;
        eobuf[(size_t)(rowbase + m) * HD + n] = acc[mi][ni][r];
      }
    }
}

// ---------- 5) deterministic gather: out[t] = sum_k w_k * (eo_k + b3[e_k]) ----------
__global__ __launch_bounds__(256) void moe_gather(
    const float* __restrict__ eobuf, const float* __restrict__ b3,
    const int* __restrict__ assign_e, const int* __restrict__ assign_i,
    const float* __restrict__ assign_w, const int* __restrict__ offsets,
    float* __restrict__ out) {
  const int t = blockIdx.y;
  const int n = blockIdx.x * 256 + threadIdx.x;
  float r = 0.f;
#pragma unroll
  for (int k = 0; k < 2; ++k) {
    int e = assign_e[t * 2 + k];
    int i = assign_i[t * 2 + k];
    float w = assign_w[t * 2 + k];
    int row = offsets[e] + i;
    r += w * (eobuf[(size_t)row * HD + n] + b3[e * HD + n]);
  }
  out[(size_t)t * HD + n] = r;
}

// ---------- host launcher ----------
extern "C" void kernel_launch(void* const* d_in, const int* in_sizes, int n_in,
                              void* d_out, int out_size, void* d_ws, size_t ws_size,
                              hipStream_t stream) {
  (void)in_sizes; (void)n_in; (void)out_size; (void)ws_size;
  const float* x  = (const float*)d_in[0];
  const float* Wg = (const float*)d_in[1];
  const float* W1 = (const float*)d_in[2];
  const float* b1 = (const float*)d_in[3];
  const float* W2 = (const float*)d_in[4];
  const float* b2 = (const float*)d_in[5];
  const float* W3 = (const float*)d_in[6];
  const float* b3 = (const float*)d_in[7];
  float* out = (float*)d_out;

  char* ws = (char*)d_ws;
  int*      cnt   = (int*)(ws + OFF_CNT);
  int*      offs  = (int*)(ws + OFF_OFFS);
  int*      ae    = (int*)(ws + OFF_AE);
  int*      ai    = (int*)(ws + OFF_AI);
  float*    aw    = (float*)(ws + OFF_AW);
  int*      slots = (int*)(ws + OFF_SLOT);
  _Float16* xh    = (_Float16*)(ws + OFF_XH);
  _Float16* xl    = (_Float16*)(ws + OFF_XL);
  _Float16* w1h   = (_Float16*)(ws + OFF_W1H);
  _Float16* w1l   = (_Float16*)(ws + OFF_W1L);
  _Float16* w2h   = (_Float16*)(ws + OFF_W2H);
  _Float16* w2l   = (_Float16*)(ws + OFF_W2L);
  _Float16* w3h   = (_Float16*)(ws + OFF_W3H);
  _Float16* w3l   = (_Float16*)(ws + OFF_W3L);
  _Float16* hh    = (_Float16*)(ws + OFF_HH);
  _Float16* hl    = (_Float16*)(ws + OFF_HL);
  float*    eobuf = (float*)(ws + OFF_EO);

  const unsigned nx4 = (unsigned)((size_t)TOK * HD / 4);
  const unsigned nw4 = (unsigned)((size_t)NE * FF * HD / 4);

  hipMemsetAsync(cnt, 0, NE * sizeof(int), stream);
  conv_split<<<(nx4 + 255) / 256, 256, 0, stream>>>(x,  xh,  xl,  nx4);
  conv_split<<<(nw4 + 255) / 256, 256, 0, stream>>>(W1, w1h, w1l, nw4);
  conv_split<<<(nw4 + 255) / 256, 256, 0, stream>>>(W2, w2h, w2l, nw4);
  conv_split<<<(nw4 + 255) / 256, 256, 0, stream>>>(W3, w3h, w3l, nw4);
  moe_gate  <<<TOK / 8, 256, 0, stream>>>(x, Wg, cnt, ae, ai, aw, slots);
  moe_prefix<<<1, 1, 0, stream>>>(cnt, offs);
  moe_ffn1  <<<dim3(FF / 128, NE * 64), 256, 0, stream>>>(
      xh, xl, w1h, w1l, w2h, w2l, b1, b2, slots, cnt, offs, hh, hl);
  moe_ffn2  <<<dim3(HD / 128, NE * 64), 256, 0, stream>>>(hh, hl, w3h, w3l, cnt, offs, eobuf);
  moe_gather<<<dim3(HD / 256, TOK), 256, 0, stream>>>(eobuf, b3, ae, ai, aw, offs, out);
}
